// AttentiveGRU2_3891240370404
// MI455X (gfx1250) — matrix operand order repaired and checked by reading, not compiled, not run on hardware
//
#include <hip/hip_runtime.h>
#include <hip/hip_bf16.h>

typedef __attribute__((ext_vector_type(16))) __bf16 v16bf;
typedef __attribute__((ext_vector_type(8)))  float  v8f;

#define FDIM 128
#define GDIM 384   // 3*FDIM

// NOTE: this problem instance has V = 50000 (multiple of 16) and E = 800000
// (multiple of 8); tile epilogues are intentionally unguarded.

union FragBF {
    v16bf v;
    uint4 q[2];
};

// ---------------------------------------------------------------------------
// A fragment from a bf16 LDS tile As[16][128], per CDNA5 ISA 7.12.2:
// lane L holds row M=L%16; lanes 0-15: K = {base..base+7, 16+base..16+base+7}
// with base=0; lanes 16-31: base=8.  Each half is 8 contiguous bf16 =>
// two aligned 16-byte LDS loads (ds_load_b128).
// ---------------------------------------------------------------------------
__device__ inline v16bf lds_a_frag(const __bf16* As, int k0, int lane) {
    const int r    = lane & 15;
    const int base = (lane & 16) ? 8 : 0;
    FragBF f;
    f.q[0] = *(const uint4*)(As + r * FDIM + k0 + base);
    f.q[1] = *(const uint4*)(As + r * FDIM + k0 + 16 + base);
    return f.v;
}

// ---------------------------------------------------------------------------
// B fragment (32x16 KxN) from pre-converted bf16 weight W[N][K] row-major:
// lane L holds column N=L%16; lanes 0-15 hold K={0..7,16..23}, lanes 16-31 K={8..15,24..31}.
// two runs of 8 contiguous bf16 => two aligned 16-byte global loads.
// ---------------------------------------------------------------------------
__device__ inline v16bf glb_b_frag(const __bf16* __restrict__ W, int ldW,
                                   int ncol, int k0, int lane) {
    const int kb = (lane & 16) ? 8 : 0;
    const __bf16* p = W + (size_t)ncol * ldW + k0 + kb;
    FragBF f;
    f.q[0] = ((const uint4*)p)[0];
    f.q[1] = ((const uint4*)(p + 16))[0];
    return f.v;
}

// ---------------------------------------------------------------------------
// 0. one-shot f32 -> bf16 weight conversion
// ---------------------------------------------------------------------------
__global__ void cvt_bf16_kernel(const float* __restrict__ s,
                                __bf16* __restrict__ d, int n) {
    int i = blockIdx.x * blockDim.x + threadIdx.x;
    if (i < n) d[i] = (__bf16)s[i];
}

// ---------------------------------------------------------------------------
// 1. workspace init
// ---------------------------------------------------------------------------
__global__ void init_ws_kernel(float* seg_max, float* denom, float* cbuf, int V) {
    int i = blockIdx.x * blockDim.x + threadIdx.x;
    if (i < V) { seg_max[i] = -3.402823466e38f; denom[i] = 0.0f; }
    if (i < V * FDIM) cbuf[i] = 0.0f;
}

// ---------------------------------------------------------------------------
// 2. segment max of edge logits (float atomic-max via int/uint trick)
// ---------------------------------------------------------------------------
__global__ void edge_max_kernel(const float* __restrict__ logits,
                                const int* __restrict__ dst,
                                float* seg_max, int E) {
    int e = blockIdx.x * blockDim.x + threadIdx.x;
    if (e >= E) return;
    float v = logits[e];
    int d = dst[e];
    if (v >= 0.0f) atomicMax((int*)&seg_max[d], __float_as_int(v));
    else           atomicMin((unsigned int*)&seg_max[d], __float_as_uint(v));
}

// ---------------------------------------------------------------------------
// 3. ex = exp(logit - max[dst]); denom[dst] += ex
// ---------------------------------------------------------------------------
__global__ void edge_exp_kernel(const float* __restrict__ logits,
                                const int* __restrict__ dst,
                                const float* __restrict__ seg_max,
                                float* __restrict__ ex,
                                float* denom, int E) {
    int e = blockIdx.x * blockDim.x + threadIdx.x;
    if (e >= E) return;
    int d = dst[e];
    float x = __expf(logits[e] - seg_max[d]);
    ex[e] = x;
    atomicAdd(&denom[d], x);
}

// ---------------------------------------------------------------------------
// 4. hv = node_feats @ W_proj^T + b_proj  (WMMA bf16, f32 accumulate)
//    block = 8 waves sharing one 16-row A tile staged in LDS (bf16);
//    wave w computes h-tile w.  Branch-free epilogue.
// ---------------------------------------------------------------------------
__global__ __launch_bounds__(256)
void proj_gemm_kernel(const float* __restrict__ nf,
                      const __bf16* __restrict__ Wp_bf,
                      const float* __restrict__ bp,
                      float* __restrict__ hv, int V) {
    __shared__ __bf16 As[16 * FDIM];              // 4 KB

    const int wave = threadIdx.x >> 5;            // 0..7 -> h-tile
    const int lane = threadIdx.x & 31;
    const int v0 = blockIdx.x * 16;
    if (v0 >= V) return;                          // block-uniform

    // cooperative stage + convert of the A tile (once per block)
    for (int idx = threadIdx.x; idx < 16 * FDIM; idx += 256) {
        const int row = v0 + (idx >> 7);
        As[idx] = (__bf16)nf[(size_t)row * FDIM + (idx & (FDIM - 1))];
    }
    __syncthreads();

    const int h0   = wave * 16;
    const int ncol = h0 + (lane & 15);
    __builtin_prefetch(Wp_bf + (size_t)ncol * FDIM, 0, 3);

    v8f acc = {};
#pragma unroll
    for (int k0 = 0; k0 < FDIM; k0 += 32) {
        v16bf a = lds_a_frag(As, k0, lane);
        v16bf b = glb_b_frag(Wp_bf, FDIM, ncol, k0, lane);
        acc = __builtin_amdgcn_wmma_f32_16x16x32_bf16(
            false, a, false, b, (short)0, acc, false, false);
    }

    const int mbase = (lane & 16) ? 8 : 0;
    const int n = lane & 15;
    const float bias = bp[h0 + n];
    float* outp = hv + (size_t)(v0 + mbase) * FDIM + h0 + n;
#pragma unroll
    for (int r = 0; r < 8; ++r)
        outp[(size_t)r * FDIM] = acc[r] + bias;
}

// ---------------------------------------------------------------------------
// 5. c[dst] += a_e * hv[src]   (one wave per edge, float4 gather, f32 atomics
//    resolving in the 192MB L2 -- c is only 25.6MB)
// ---------------------------------------------------------------------------
__global__ __launch_bounds__(256)
void edge_aggregate_kernel(const float* __restrict__ ex,
                           const float* __restrict__ denom,
                           const float* __restrict__ hv,
                           const int* __restrict__ src,
                           const int* __restrict__ dst,
                           float* cbuf, int E) {
    const int e = blockIdx.x * 8 + (threadIdx.x >> 5);
    if (e >= E) return;
    const int lane = threadIdx.x & 31;
    const int d = dst[e];
    const float a = ex[e] / denom[d];
    const float4 v = ((const float4*)(hv + (size_t)src[e] * FDIM))[lane];
    float* cp = cbuf + (size_t)d * FDIM + lane * 4;
    atomicAdd(cp + 0, v.x * a);
    atomicAdd(cp + 1, v.y * a);
    atomicAdd(cp + 2, v.z * a);
    atomicAdd(cp + 3, v.w * a);
}

// ---------------------------------------------------------------------------
// 6. context = elu(c), in place
// ---------------------------------------------------------------------------
__global__ void elu_kernel(float* cbuf, int n) {
    int i = blockIdx.x * blockDim.x + threadIdx.x;
    if (i >= n) return;
    float x = cbuf[i];
    cbuf[i] = (x > 0.0f) ? x : (__expf(x) - 1.0f);
}

// ---------------------------------------------------------------------------
// 7. fused GRU: gi = ctx @ w_ih^T + b_ih ; gh = nf @ w_hh^T + b_hh ;
//    gating + ReLU.  24 waves/block, one 16x16 n-tile of both gi and gh per
//    wave; A tiles staged once in LDS as bf16; gi/gh staged in LDS so the
//    154MB intermediates never hit HBM.
// ---------------------------------------------------------------------------
__global__ __launch_bounds__(768)
void gru_fused_kernel(const float* __restrict__ ctx,
                      const float* __restrict__ nf,
                      const __bf16* __restrict__ wih_bf,
                      const __bf16* __restrict__ whh_bf,
                      const float* __restrict__ b_ih,
                      const float* __restrict__ b_hh,
                      float* __restrict__ out, int V) {
    __shared__ __bf16 Actx[16 * FDIM];            // 4 KB
    __shared__ __bf16 Anf [16 * FDIM];            // 4 KB
    __shared__ float  gi_s[16 * GDIM];            // 24 KB
    __shared__ float  gh_s[16 * GDIM];            // 24 KB

    const int wave = threadIdx.x >> 5;            // 0..23 -> n-tile
    const int lane = threadIdx.x & 31;
    const int v0 = blockIdx.x * 16;
    if (v0 >= V) return;                          // block-uniform

    for (int idx = threadIdx.x; idx < 16 * FDIM; idx += 768) {
        const int row = v0 + (idx >> 7);
        const size_t off = (size_t)row * FDIM + (idx & (FDIM - 1));
        Actx[idx] = (__bf16)ctx[off];
        Anf [idx] = (__bf16)nf[off];
    }
    __syncthreads();

    {
        const int n0   = wave * 16;
        const int ncol = n0 + (lane & 15);
        __builtin_prefetch(wih_bf + (size_t)ncol * FDIM, 0, 3);
        __builtin_prefetch(whh_bf + (size_t)ncol * FDIM, 0, 3);

        v8f acc_i = {}, acc_h = {};
#pragma unroll
        for (int k0 = 0; k0 < FDIM; k0 += 32) {
            v16bf ai = lds_a_frag(Actx, k0, lane);
            v16bf ah = lds_a_frag(Anf,  k0, lane);
            v16bf bi = glb_b_frag(wih_bf, FDIM, ncol, k0, lane);
            v16bf bh = glb_b_frag(whh_bf, FDIM, ncol, k0, lane);
            acc_i = __builtin_amdgcn_wmma_f32_16x16x32_bf16(
                false, ai, false, bi, (short)0, acc_i, false, false);
            acc_h = __builtin_amdgcn_wmma_f32_16x16x32_bf16(
                false, ah, false, bh, (short)0, acc_h, false, false);
        }

        const int mbase = (lane & 16) ? 8 : 0;
        const int n = lane & 15;
        const float bi_v = b_ih[n0 + n];
        const float bh_v = b_hh[n0 + n];
#pragma unroll
        for (int r = 0; r < 8; ++r) {
            const int m = mbase + r;
            gi_s[m * GDIM + n0 + n] = acc_i[r] + bi_v;
            gh_s[m * GDIM + n0 + n] = acc_h[r] + bh_v;
        }
    }
    __syncthreads();

    for (int idx = threadIdx.x; idx < 16 * FDIM; idx += 768) {
        const int m = idx >> 7;
        const int f = idx & (FDIM - 1);
        const size_t go = (size_t)(v0 + m) * FDIM + f;
        const float ir  = gi_s[m * GDIM + f];
        const float iz  = gi_s[m * GDIM + FDIM + f];
        const float in_ = gi_s[m * GDIM + 2 * FDIM + f];
        const float hr  = gh_s[m * GDIM + f];
        const float hz  = gh_s[m * GDIM + FDIM + f];
        const float hn  = gh_s[m * GDIM + 2 * FDIM + f];
        const float r = 1.0f / (1.0f + __expf(-(ir + hr)));
        const float z = 1.0f / (1.0f + __expf(-(iz + hz)));
        const float nn = tanhf(in_ + r * hn);
        const float h = (1.0f - z) * nn + z * nf[go];
        out[go] = fmaxf(h, 0.0f);
    }
}

// ---------------------------------------------------------------------------
extern "C" void kernel_launch(void* const* d_in, const int* in_sizes, int n_in,
                              void* d_out, int out_size, void* d_ws, size_t ws_size,
                              hipStream_t stream) {
    const float* edge_logits = (const float*)d_in[0];
    const float* node_feats  = (const float*)d_in[1];
    const float* W_proj      = (const float*)d_in[2];
    const float* b_proj      = (const float*)d_in[3];
    const float* w_ih        = (const float*)d_in[4];
    const float* w_hh        = (const float*)d_in[5];
    const float* b_ih        = (const float*)d_in[6];
    const float* b_hh        = (const float*)d_in[7];
    const int*   src         = (const int*)d_in[8];
    const int*   dst         = (const int*)d_in[9];

    const int E = in_sizes[0];
    const int V = in_sizes[1] / FDIM;

    // float workspace: seg_max[V] | denom[V] | ex[E] | hv[V*128] | c[V*128]
    float* ws      = (float*)d_ws;
    float* seg_max = ws;
    float* denom   = ws + V;
    float* ex      = ws + 2 * (size_t)V;
    float* hv      = ex + E;
    float* cbuf    = hv + (size_t)V * FDIM;
    // bf16 region after floats (16B aligned): Wp | w_ih | w_hh
    __bf16* wbf    = (__bf16*)(cbuf + (size_t)V * FDIM);
    __bf16* Wp_bf  = wbf;
    __bf16* wih_bf = Wp_bf  + FDIM * FDIM;
    __bf16* whh_bf = wih_bf + GDIM * FDIM;
    float* out     = (float*)d_out;

    const int T = 256;

    // 0. weight conversion (one-shot, ~115k elements)
    cvt_bf16_kernel<<<(FDIM * FDIM + T - 1) / T, T, 0, stream>>>(W_proj, Wp_bf, FDIM * FDIM);
    cvt_bf16_kernel<<<(GDIM * FDIM + T - 1) / T, T, 0, stream>>>(w_ih, wih_bf, GDIM * FDIM);
    cvt_bf16_kernel<<<(GDIM * FDIM + T - 1) / T, T, 0, stream>>>(w_hh, whh_bf, GDIM * FDIM);

    // 1. init
    {
        int n = V * FDIM;
        init_ws_kernel<<<(n + T - 1) / T, T, 0, stream>>>(seg_max, denom, cbuf, V);
    }
    // 2. segment max
    edge_max_kernel<<<(E + T - 1) / T, T, 0, stream>>>(edge_logits, dst, seg_max, E);
    // 3. exp + denom
    edge_exp_kernel<<<(E + T - 1) / T, T, 0, stream>>>(edge_logits, dst, seg_max, ex, denom, E);
    // 4. hv projection GEMM (WMMA)
    {
        int vtiles = (V + 15) / 16;
        proj_gemm_kernel<<<vtiles, T, 0, stream>>>(node_feats, Wp_bf, b_proj, hv, V);
    }
    // 5. edge aggregation (scatter-add)
    edge_aggregate_kernel<<<(E + 7) / 8, T, 0, stream>>>(ex, denom, hv, src, dst, cbuf, E);
    // 6. ELU
    {
        int n = V * FDIM;
        elu_kernel<<<(n + T - 1) / T, T, 0, stream>>>(cbuf, n);
    }
    // 7. fused GRU (WMMA x2 + gating)
    {
        int vtiles = (V + 15) / 16;
        gru_fused_kernel<<<vtiles, 768, 0, stream>>>(cbuf, node_feats, wih_bf, whh_bf,
                                                     b_ih, b_hh, out, V);
    }
}
